// MultiHeadLatentAttention_54769422958704
// MI455X (gfx1250) — matrix operand order repaired and checked by reading, not compiled, not run on hardware
//
#include <hip/hip_runtime.h>

#define HH    16
#define TT    2048
#define BBATCH 2
#define DDIM  2048
#define NOPE  128
#define ROPE_D 64
#define QKD   192
#define KVR   512
#define VD    128

typedef __attribute__((ext_vector_type(16))) __bf16 bf16x16;
typedef __attribute__((ext_vector_type(8)))  float  f32x8;
typedef __attribute__((ext_vector_type(4)))  float  f32x4;
typedef __attribute__((ext_vector_type(4)))  unsigned int u32x4;
typedef __attribute__((ext_vector_type(2)))  unsigned int u32x2;

union Frag16 { bf16x16 v; u32x4 q[2]; };

__device__ __forceinline__ unsigned short f2bf(float f) {
    unsigned int u = __float_as_uint(f);
    u += 0x7FFFu + ((u >> 16) & 1u);           // round-to-nearest-even
    return (unsigned short)(u >> 16);
}
__device__ __forceinline__ unsigned int pack_bf2(float a, float b) {
    return (unsigned int)f2bf(a) | ((unsigned int)f2bf(b) << 16);
}
__device__ __forceinline__ f32x8 wmma_bf16(const Frag16& a, const Frag16& b, f32x8 c) {
    return __builtin_amdgcn_wmma_f32_16x16x32_bf16(false, a.v, false, b.v,
                                                   (short)0, c, false, false);
}

// ---------------------------------------------------------------------------
// Generic tiled GEMM: C[M,N] = A[M,K] * B[K,N], f32 in, bf16 WMMA, f32 acc.
// Block 128x128, BK=32, 256 threads = 8 waves (4 along M x 2 along N).
// MODE 0: plain f32 store to Cout.  MODE 1: kv_b epilogue -> Kf/Vf (bf16).
// ---------------------------------------------------------------------------
template<int MODE>
__global__ __launch_bounds__(256) void gemm_wmma_bf16(
    const float* __restrict__ A, int lda,
    const float* __restrict__ Bw, int ldb,
    int Ncols, int Kdim,
    float* __restrict__ Cout, int ldc,
    unsigned short* __restrict__ Kf, unsigned short* __restrict__ Vf)
{
    __shared__ __align__(16) unsigned short As[128 * 32];
    __shared__ __align__(16) unsigned short Bs[128 * 32];
    const int tid  = threadIdx.x;
    const int lane = tid & 31;
    const int wave = tid >> 5;
    const int wm   = wave & 3;     // 32-row slice
    const int wn   = wave >> 2;    // 64-col slice
    const int m0   = blockIdx.y * 128;
    const int n0   = blockIdx.x * 128;
    const int l16  = lane & 15;
    const int hs   = lane >> 4;

    f32x8 acc[2][4];
    #pragma unroll
    for (int i = 0; i < 2; ++i)
        #pragma unroll
        for (int j = 0; j < 4; ++j)
            #pragma unroll
            for (int v = 0; v < 8; ++v) acc[i][j][v] = 0.f;

    for (int k0 = 0; k0 < Kdim; k0 += 32) {
        __syncthreads();
        // stage A tile 128x32, f32 -> bf16, row-major (64B rows)
        #pragma unroll
        for (int c = 0; c < 4; ++c) {
            int e   = tid + 256 * c;          // 0..1023
            int row = e >> 3;
            int kq  = (e & 7) << 2;
            f32x4 v = *(const f32x4*)(A + (size_t)(m0 + row) * lda + k0 + kq);
            u32x2 p; p.x = pack_bf2(v.x, v.y); p.y = pack_bf2(v.z, v.w);
            *(u32x2*)&As[row * 32 + kq] = p;
        }
        // stage B tile 32x128, f32 -> bf16, transposed (K contiguous per column)
        #pragma unroll
        for (int c = 0; c < 4; ++c) {
            int e  = tid + 256 * c;
            int kr = e >> 5;
            int nq = (e & 31) << 2;
            f32x4 v = {0.f, 0.f, 0.f, 0.f};
            if (n0 + nq < Ncols)
                v = *(const f32x4*)(Bw + (size_t)(k0 + kr) * ldb + n0 + nq);
            Bs[(nq + 0) * 32 + kr] = f2bf(v.x);
            Bs[(nq + 1) * 32 + kr] = f2bf(v.y);
            Bs[(nq + 2) * 32 + kr] = f2bf(v.z);
            Bs[(nq + 3) * 32 + kr] = f2bf(v.w);
        }
        __syncthreads();

        Frag16 af[2], bfr[4];
        #pragma unroll
        for (int mi = 0; mi < 2; ++mi) {
            int row = wm * 32 + mi * 16 + l16;
            int s0  = row * 32 + hs * 8;       // lanes<16: K0-7 & 16-23
            af[mi].q[0] = *(const u32x4*)&As[s0];
            af[mi].q[1] = *(const u32x4*)&As[s0 + 16];
        }
        #pragma unroll
        for (int ni = 0; ni < 4; ++ni) {
            int col = wn * 64 + ni * 16 + l16;
            int s0  = col * 32 + hs * 8;       // lanes<16: K0-7 & 16-23
            bfr[ni].q[0] = *(const u32x4*)&Bs[s0];
            bfr[ni].q[1] = *(const u32x4*)&Bs[s0 + 16];
        }
        #pragma unroll
        for (int mi = 0; mi < 2; ++mi)
            #pragma unroll
            for (int ni = 0; ni < 4; ++ni)
                acc[mi][ni] = wmma_bf16(af[mi], bfr[ni], acc[mi][ni]);
    }

    // epilogue (C layout: lane = N%16, VGPR v = M row, halves offset by 8)
    #pragma unroll
    for (int mi = 0; mi < 2; ++mi) {
        #pragma unroll
        for (int ni = 0; ni < 4; ++ni) {
            int col = n0 + wn * 64 + ni * 16 + l16;
            if (col >= Ncols) continue;
            #pragma unroll
            for (int v = 0; v < 8; ++v) {
                int row = m0 + wm * 32 + mi * 16 + hs * 8 + v;
                float val = acc[mi][ni][v];
                if (MODE == 0) {
                    Cout[(size_t)row * ldc + col] = val;
                } else {
                    int b = row / TT, t = row % TT;
                    int h = col >> 8, d = col & 255;
                    size_t base = (size_t)(b * HH + h) * TT + t;
                    if (d < NOPE) Kf[base * QKD + d]        = f2bf(val);
                    else          Vf[base * VD + (d - NOPE)] = f2bf(val);
                }
            }
        }
    }
}

// ---------------------------------------------------------------------------
// RoPE on q: q_raw [B,T,H*192] f32 -> Qf [B,H,T,192] bf16 (rope on last 64)
// ---------------------------------------------------------------------------
__global__ void rope_q_kernel(const float* __restrict__ qraw,
                              unsigned short* __restrict__ Qf)
{
    int idx = blockIdx.x;                   // (b*TT + t)*HH + h
    int h = idx % HH;
    int t = (idx / HH) % TT;
    int b = idx / (HH * TT);
    int d = threadIdx.x;                    // 0..191
    const float* src = qraw + ((size_t)(b * TT + t)) * (HH * QKD) + h * QKD;
    float x = src[d];
    float outv;
    if (d < NOPE) {
        outv = x;
    } else {
        int i = d - NOPE;                   // 0..63
        int j = i & 31;
        float freq = (float)t * __powf(10000.0f, -(float)j * (1.0f / 32.0f));
        float s, c; __sincosf(freq, &s, &c);
        float pair = src[NOPE + ((i < 32) ? i + 32 : i - 32)];
        outv = x * c + ((i < 32) ? -pair : pair) * s;
    }
    Qf[((size_t)(b * HH + h) * TT + t) * QKD + d] = f2bf(outv);
}

// ---------------------------------------------------------------------------
// Per-token: RMS-norm first 512 of kvc in place (latent), RoPE last 64 and
// broadcast to all heads' K rope slots.
// ---------------------------------------------------------------------------
__global__ __launch_bounds__(256) void kv_fill_kernel(
    float* __restrict__ kvc, const float* __restrict__ w,
    unsigned short* __restrict__ Kf)
{
    __shared__ float red[8];
    int token = blockIdx.x;
    int b = token / TT, t = token % TT;
    float* row = kvc + (size_t)token * (KVR + ROPE_D);
    int tid = threadIdx.x;
    float x0 = row[tid], x1 = row[tid + 256];
    float ss = x0 * x0 + x1 * x1;
    #pragma unroll
    for (int m = 1; m < 32; m <<= 1) ss += __shfl_xor(ss, m, 32);
    if ((tid & 31) == 0) red[tid >> 5] = ss;
    __syncthreads();
    float tot = red[0] + red[1] + red[2] + red[3] + red[4] + red[5] + red[6] + red[7];
    float nrm = rsqrtf(tot * (1.0f / 512.0f) + 1e-6f);
    row[tid]       = x0 * nrm * w[tid];
    row[tid + 256] = x1 * nrm * w[tid + 256];
    if (tid < ROPE_D) {
        int i = tid;
        float x    = row[KVR + i];
        float pair = row[KVR + ((i < 32) ? i + 32 : i - 32)];
        int j = i & 31;
        float freq = (float)t * __powf(10000.0f, -(float)j * (1.0f / 32.0f));
        float s, c; __sincosf(freq, &s, &c);
        unsigned short kb = f2bf(x * c + ((i < 32) ? -pair : pair) * s);
        #pragma unroll
        for (int h = 0; h < HH; ++h)
            Kf[((size_t)(b * HH + h) * TT + t) * QKD + NOPE + i] = kb;
    }
}

// ---------------------------------------------------------------------------
// Causal flash attention. Grid (T/128, B*H), 256 threads = 8 waves, each
// wave owns 16 queries; 32-key blocks staged in LDS and shared by all waves.
// ---------------------------------------------------------------------------
__global__ __launch_bounds__(256) void mla_flash_attn(
    const unsigned short* __restrict__ Qf,
    const unsigned short* __restrict__ Kf,
    const unsigned short* __restrict__ Vf,
    float* __restrict__ Y)
{
    __shared__ __align__(16) unsigned short Ks[32 * QKD];   // 12KB row-major
    __shared__ __align__(16) unsigned short Vt[VD * 32];    // 8KB transposed
    __shared__ __align__(16) unsigned short Ps[8 * 16 * 32];// 8KB per-wave P

    const int tid  = threadIdx.x;
    const int lane = tid & 31;
    const int wave = tid >> 5;
    const int l16  = lane & 15;
    const int hs   = lane >> 4;
    const int bh   = blockIdx.y;
    const int q0   = blockIdx.x * 128 + wave * 16;
    const float scale = 0.07216878364870323f;  // 192^-0.5

    // Q A-fragments directly from global [B,H,T,192] bf16
    Frag16 qf[6];
    {
        const unsigned short* qrow = Qf + ((size_t)bh * TT + (q0 + l16)) * QKD;
        #pragma unroll
        for (int c = 0; c < 6; ++c) {
            int s0 = c * 32 + hs * 8;
            qf[c].q[0] = *(const u32x4*)(qrow + s0);
            qf[c].q[1] = *(const u32x4*)(qrow + s0 + 16);
        }
    }

    float m_i[8], l_i[8];
    f32x8 o[8];
    #pragma unroll
    for (int v = 0; v < 8; ++v) { m_i[v] = -1e30f; l_i[v] = 0.f; }
    #pragma unroll
    for (int nt = 0; nt < 8; ++nt)
        #pragma unroll
        for (int v = 0; v < 8; ++v) o[nt][v] = 0.f;

    unsigned short* Pw = Ps + wave * 512;
    const int kv_end = blockIdx.x * 128 + 128;

    for (int kv0 = 0; kv0 < kv_end; kv0 += 32) {
        // stage K rows [kv0,kv0+32): straight 12KB copy (bf16 already)
        #pragma unroll
        for (int c = 0; c < 3; ++c) {
            int e   = tid + 256 * c;   // 0..767 uint4 chunks
            int key = e / 24, ch = e % 24;
            ((u32x4*)Ks)[key * 24 + ch] =
                ((const u32x4*)(Kf + ((size_t)bh * TT + kv0 + key) * QKD))[ch];
        }
        // stage V transposed: Vt[dim][key]
        #pragma unroll
        for (int c = 0; c < 4; ++c) {
            int e   = tid + 256 * c;   // 0..1023
            int key = e >> 5;
            int d4  = (e & 31) << 2;
            union { u32x2 u; unsigned short s[4]; } r;
            r.u = *(const u32x2*)(Vf + ((size_t)bh * TT + kv0 + key) * VD + d4);
            Vt[(d4 + 0) * 32 + key] = r.s[0];
            Vt[(d4 + 1) * 32 + key] = r.s[1];
            Vt[(d4 + 2) * 32 + key] = r.s[2];
            Vt[(d4 + 3) * 32 + key] = r.s[3];
        }
        if (kv0 + 32 < kv_end)
            __builtin_prefetch(Kf + ((size_t)bh * TT + kv0 + 32) * QKD, 0, 0);
        __syncthreads();

        if (kv0 <= q0 + 15) {          // wave-uniform: EXEC stays all-ones
            f32x8 s0a, s1a;
            #pragma unroll
            for (int v = 0; v < 8; ++v) { s0a[v] = 0.f; s1a[v] = 0.f; }
            // S = Q K^T : 12 WMMAs
            #pragma unroll
            for (int c = 0; c < 6; ++c) {
                int k0off = l16 * QKD + c * 32 + hs * 8;
                Frag16 kfa; kfa.q[0] = *(const u32x4*)&Ks[k0off];
                            kfa.q[1] = *(const u32x4*)&Ks[k0off + 16];
                s0a = wmma_bf16(qf[c], kfa, s0a);
                int k1off = (16 + l16) * QKD + c * 32 + hs * 8;
                Frag16 kfb; kfb.q[0] = *(const u32x4*)&Ks[k1off];
                            kfb.q[1] = *(const u32x4*)&Ks[k1off + 16];
                s1a = wmma_bf16(qf[c], kfb, s1a);
            }
            // online softmax (rows: lane<16 -> 0..7, lane>=16 -> 8..15)
            float p0v[8], p1v[8];
            #pragma unroll
            for (int v = 0; v < 8; ++v) {
                int row = q0 + hs * 8 + v;
                float x0 = s0a[v] * scale;
                float x1 = s1a[v] * scale;
                if (kv0 + l16 > row)      x0 = -1e30f;
                if (kv0 + 16 + l16 > row) x1 = -1e30f;
                float rm = fmaxf(x0, x1);
                rm = fmaxf(rm, __shfl_xor(rm, 1, 32));
                rm = fmaxf(rm, __shfl_xor(rm, 2, 32));
                rm = fmaxf(rm, __shfl_xor(rm, 4, 32));
                rm = fmaxf(rm, __shfl_xor(rm, 8, 32));
                float mn = fmaxf(m_i[v], rm);
                float al = __expf(m_i[v] - mn);
                float p0 = __expf(x0 - mn);
                float p1 = __expf(x1 - mn);
                float rs = p0 + p1;
                rs += __shfl_xor(rs, 1, 32);
                rs += __shfl_xor(rs, 2, 32);
                rs += __shfl_xor(rs, 4, 32);
                rs += __shfl_xor(rs, 8, 32);
                l_i[v] = l_i[v] * al + rs;
                m_i[v] = mn;
                #pragma unroll
                for (int nt = 0; nt < 8; ++nt) o[nt][v] *= al;
                p0v[v] = p0; p1v[v] = p1;
            }
            // P (C layout) -> LDS -> A-fragment layout
            #pragma unroll
            for (int v = 0; v < 8; ++v) {
                int rl = hs * 8 + v;
                Pw[rl * 32 + l16]      = f2bf(p0v[v]);
                Pw[rl * 32 + 16 + l16] = f2bf(p1v[v]);
            }
            asm volatile("s_wait_dscnt 0" ::: "memory");
            Frag16 pf;
            { int s0 = l16 * 32 + hs * 8;
              pf.q[0] = *(const u32x4*)&Pw[s0];
              pf.q[1] = *(const u32x4*)&Pw[s0 + 16]; }
            // O += P @ V : 8 WMMAs into 16x128 accumulator
            #pragma unroll
            for (int nt = 0; nt < 8; ++nt) {
                int vo = (nt * 16 + l16) * 32 + hs * 8;
                Frag16 vfr; vfr.q[0] = *(const u32x4*)&Vt[vo];
                            vfr.q[1] = *(const u32x4*)&Vt[vo + 16];
                o[nt] = wmma_bf16(pf, vfr, o[nt]);
            }
        }
        __syncthreads();
    }

    const int b = bh / HH, h = bh % HH;
    #pragma unroll
    for (int v = 0; v < 8; ++v) {
        int row = q0 + hs * 8 + v;
        float inv = 1.0f / l_i[v];
        float* yr = Y + ((size_t)(b * TT + row)) * (HH * VD) + h * VD;
        #pragma unroll
        for (int nt = 0; nt < 8; ++nt)
            yr[nt * 16 + l16] = o[nt][v] * inv;
    }
}

// ---------------------------------------------------------------------------
extern "C" void kernel_launch(void* const* d_in, const int* in_sizes, int n_in,
                              void* d_out, int out_size, void* d_ws, size_t ws_size,
                              hipStream_t stream)
{
    const float* x     = (const float*)d_in[0];
    const float* wq    = (const float*)d_in[1];
    const float* wkv_a = (const float*)d_in[2];
    const float* wkv_b = (const float*)d_in[3];
    const float* wo    = (const float*)d_in[4];
    const float* kvw   = (const float*)d_in[5];
    float* out = (float*)d_out;

    const int M = BBATCH * TT;                  // 4096 tokens
    char* ws = (char*)d_ws;
    size_t off = 0;
    auto alloc = [&](size_t bytes) {
        void* p = ws + off;
        off += (bytes + 255) & ~(size_t)255;
        return p;
    };
    float*          q_raw = (float*)alloc((size_t)M * (HH * QKD) * 4);  // reused as Y
    unsigned short* Qf    = (unsigned short*)alloc((size_t)BBATCH * HH * TT * QKD * 2);
    float*          kvc   = (float*)alloc((size_t)M * (KVR + ROPE_D) * 4);
    unsigned short* Kf    = (unsigned short*)alloc((size_t)BBATCH * HH * TT * QKD * 2);
    unsigned short* Vf    = (unsigned short*)alloc((size_t)BBATCH * HH * TT * VD * 2);
    float* Y = q_raw;  // q_raw is dead after rope_q; reuse for attention output

    dim3 blk(256);
    // 1. q = x @ wq          (4096 x 2048 x 3072)
    gemm_wmma_bf16<0><<<dim3(24, 32), blk, 0, stream>>>(
        x, DDIM, wq, HH * QKD, HH * QKD, DDIM, q_raw, HH * QKD, nullptr, nullptr);
    // 2. kvc = x @ wkv_a     (4096 x 2048 x 576)
    gemm_wmma_bf16<0><<<dim3(5, 32), blk, 0, stream>>>(
        x, DDIM, wkv_a, KVR + ROPE_D, KVR + ROPE_D, DDIM, kvc, KVR + ROPE_D, nullptr, nullptr);
    // 3. RoPE(q) -> Qf bf16 [B,H,T,192]
    rope_q_kernel<<<dim3(M * HH), dim3(QKD), 0, stream>>>(q_raw, Qf);
    // 4. rmsnorm latent in place + RoPE(k_rope) -> Kf rope slots
    kv_fill_kernel<<<dim3(M), blk, 0, stream>>>(kvc, kvw, Kf);
    // 5. kv_full = latent @ wkv_b (4096 x 512 x 4096) -> Kf nope / Vf
    gemm_wmma_bf16<1><<<dim3(32, 32), blk, 0, stream>>>(
        kvc, KVR + ROPE_D, wkv_b, HH * (NOPE + VD), HH * (NOPE + VD), KVR,
        nullptr, 0, Kf, Vf);
    // 6. causal flash attention -> Y f32 [B,T,2048]
    mla_flash_attn<<<dim3(TT / 128, BBATCH * HH), blk, 0, stream>>>(Qf, Kf, Vf, Y);
    // 7. out = Y @ wo        (4096 x 2048 x 2048)
    gemm_wmma_bf16<0><<<dim3(16, 32), blk, 0, stream>>>(
        Y, HH * VD, wo, DDIM, DDIM, HH * VD, out, DDIM, nullptr, nullptr);
}
